// OctoTransformer_31825707663649
// MI455X (gfx1250) — matrix operand order repaired and checked by reading, not compiled, not run on hardware
//
#include <hip/hip_runtime.h>
#include <cmath>

typedef __attribute__((ext_vector_type(16))) _Float16 v16h;
typedef __attribute__((ext_vector_type(8)))  _Float16 v8h;
typedef __attribute__((ext_vector_type(8)))  float    v8f;

#define NBATCH 128
#define TOK    291
#define DEMB   384
#define NHEAD  6
#define HDIM   64
#define NLAYER 12
#define MROWS  (NBATCH * TOK)   // 37248 = 291 * 128

__device__ __forceinline__ v8f wmma32(v16h a, v16h b, v8f c) {
  return __builtin_amdgcn_wmma_f32_16x16x32_f16(false, a, false, b, (short)0, c, false, false);
}

// A-fragment pattern: 16 contiguous halves from p (16-B aligned).
__device__ __forceinline__ v16h ldfragA(const _Float16* p) {
  v8h lo = *(const v8h*)p;
  v8h hi = *(const v8h*)(p + 8);
  return __builtin_shufflevector(lo, hi, 0,1,2,3,4,5,6,7,8,9,10,11,12,13,14,15);
}
// 16 contiguous halves from p (16-B aligned).
__device__ __forceinline__ v16h ldfrag16(const _Float16* p) {
  v8h lo = *(const v8h*)p;
  v8h hi = *(const v8h*)(p + 8);
  return __builtin_shufflevector(lo, hi, 0,1,2,3,4,5,6,7,8,9,10,11,12,13,14,15);
}

__device__ __forceinline__ float fast_gelu(float x) {
  // 0.5x(1+tanh(u)) == x * sigmoid(2u),  u = 0.7978845608(x + 0.044715 x^3)
  float z = 1.5957691216f * (x + 0.044715f * x * x * x);
  return x * __builtin_amdgcn_rcpf(1.0f + __expf(-z));
}

// ---------------------------------------------------------------------------
// WMMA GEMM:  C[M x N] = act(A_f16[M x K] @ B_f32[K x N] + bias)
// block tile 128x64, 8 waves (4x2), wave tile 32x32, K step 32 through LDS.
// B staged TRANSPOSED in LDS so B-fragments are contiguous b128 loads.
// flags: bit0 = tanh-GELU, bit1 = residual add into outF.
// ---------------------------------------------------------------------------
__global__ void __launch_bounds__(256)
gemm_kernel(const _Float16* __restrict__ A, const float* __restrict__ Bw,
            const float* __restrict__ bias, float* __restrict__ outF,
            _Float16* __restrict__ outH, int K, int N, int flags) {
  __shared__ __align__(16) _Float16 As[128 * 40];   // [row][k], stride 40 (80B)
  __shared__ __align__(16) _Float16 BsT[64 * 40];   // [col][k], stride 40 (80B)

  const int tid  = threadIdx.x;
  const int n0   = blockIdx.x * 64;
  const int m0   = blockIdx.y * 128;
  const int lane = tid & 31;
  const int wave = tid >> 5;
  const int wm   = wave >> 1;          // 0..3
  const int wn   = wave & 1;           // 0..1
  const int grp  = lane >> 4;
  const int ln   = lane & 15;

  v8f zero = {};
  v8f c00 = zero, c01 = zero, c10 = zero, c11 = zero;

  for (int k0 = 0; k0 < K; k0 += 32) {
    __syncthreads();
    // stage A tile: 128x32 halves = 512 v8h chunks, 2 per thread
    #pragma unroll
    for (int c = 0; c < 2; ++c) {
      int idx = tid + c * 256;
      int r = idx >> 2, c8 = (idx & 3) * 8;
      *(v8h*)&As[r * 40 + c8] = *(const v8h*)&A[(size_t)(m0 + r) * K + k0 + c8];
    }
    // stage B tile transposed: 32x64 f32 = 512 float4 chunks, 2 per thread
    #pragma unroll
    for (int c = 0; c < 2; ++c) {
      int idx = tid + c * 256;
      int r = idx >> 4, c4 = (idx & 15) * 4;
      float4 f = *(const float4*)&Bw[(size_t)(k0 + r) * N + n0 + c4];
      BsT[(c4 + 0) * 40 + r] = (_Float16)f.x;
      BsT[(c4 + 1) * 40 + r] = (_Float16)f.y;
      BsT[(c4 + 2) * 40 + r] = (_Float16)f.z;
      BsT[(c4 + 3) * 40 + r] = (_Float16)f.w;
    }
    __syncthreads();

    v16h a0 = ldfragA(&As[(wm * 32 + 0  + ln) * 40 + grp * 16]);
    v16h a1 = ldfragA(&As[(wm * 32 + 16 + ln) * 40 + grp * 16]);
    v16h b0 = ldfrag16(&BsT[(wn * 32 + 0  + ln) * 40 + grp * 16]);
    v16h b1 = ldfrag16(&BsT[(wn * 32 + 16 + ln) * 40 + grp * 16]);

    c00 = wmma32(a0, b0, c00);
    c01 = wmma32(a0, b1, c01);
    c10 = wmma32(a1, b0, c10);
    c11 = wmma32(a1, b1, c11);
  }

  // epilogue: uniform branches hoisted outside the element loops
  v8f cf[2][2];
  cf[0][0] = c00; cf[0][1] = c01; cf[1][0] = c10; cf[1][1] = c11;
  const int colb = n0 + wn * 32 + ln;
  const int rowb = m0 + wm * 32 + grp * 8;

  if (outH) {
    if (flags & 1) {
      #pragma unroll
      for (int ms = 0; ms < 2; ++ms)
        #pragma unroll
        for (int ns = 0; ns < 2; ++ns) {
          int col = colb + ns * 16;
          float bv = bias[col];
          #pragma unroll
          for (int i = 0; i < 8; ++i) {
            size_t o = (size_t)(rowb + ms * 16 + i) * N + col;
            outH[o] = (_Float16)fast_gelu(cf[ms][ns][i] + bv);
          }
        }
    } else {
      #pragma unroll
      for (int ms = 0; ms < 2; ++ms)
        #pragma unroll
        for (int ns = 0; ns < 2; ++ns) {
          int col = colb + ns * 16;
          float bv = bias[col];
          #pragma unroll
          for (int i = 0; i < 8; ++i) {
            size_t o = (size_t)(rowb + ms * 16 + i) * N + col;
            outH[o] = (_Float16)(cf[ms][ns][i] + bv);
          }
        }
    }
  } else if (flags & 2) {
    #pragma unroll
    for (int ms = 0; ms < 2; ++ms)
      #pragma unroll
      for (int ns = 0; ns < 2; ++ns) {
        int col = colb + ns * 16;
        float bv = bias[col];
        #pragma unroll
        for (int i = 0; i < 8; ++i) {
          size_t o = (size_t)(rowb + ms * 16 + i) * N + col;
          outF[o] += cf[ms][ns][i] + bv;
        }
      }
  } else {
    #pragma unroll
    for (int ms = 0; ms < 2; ++ms)
      #pragma unroll
      for (int ns = 0; ns < 2; ++ns) {
        int col = colb + ns * 16;
        float bv = bias[col];
        #pragma unroll
        for (int i = 0; i < 8; ++i) {
          size_t o = (size_t)(rowb + ms * 16 + i) * N + col;
          outF[o] = cf[ms][ns][i] + bv;
        }
      }
  }
}

// ---------------------------------------------------------------------------
// WMMA flash attention: 1 wave per (batch, head, 16-query block).
// qkv: f16 [M x 1152] rows b*291+t, cols [Q | K | V], 6 heads x 64.
// Mask: keys >= 291 invalid; query 290 (readout) attends only to key 290.
// ---------------------------------------------------------------------------
__global__ void __launch_bounds__(32)
attn_kernel(const _Float16* __restrict__ qkv, _Float16* __restrict__ out) {
  int id   = blockIdx.x;
  int qb   = id % 19;  id /= 19;
  int head = id % NHEAD;
  int b    = id / NHEAD;
  const int lane = threadIdx.x;
  const int grp  = lane >> 4;
  const int ln   = lane & 15;
  const float NEG = -1e30f;

  const size_t base = (size_t)b * TOK * 1152;

  // Q fragments (A layout); fold 1/sqrt(64) = 0.125 into Q (exact in f16)
  int qr = qb * 16 + ln; if (qr > 290) qr = 290;
  const _Float16* qp = qkv + base + (size_t)qr * 1152 + head * HDIM + grp * 16;
  v16h aq0 = ldfragA(qp)      * (_Float16)0.125f;
  v16h aq1 = ldfragA(qp + 32) * (_Float16)0.125f;

  float m[8], lsum[8], sf[8];
  v8f zero = {};
  v8f acc0 = zero, acc1 = zero, acc2 = zero, acc3 = zero;
  #pragma unroll
  for (int i = 0; i < 8; ++i) { m[i] = NEG; lsum[i] = 0.f; }

  __shared__ __align__(16) _Float16 Pl[16 * 32];    // P tile, row-major
  __shared__ __align__(16) _Float16 VsT[64 * 40];   // V transposed: [d][key]

  for (int jj = 0; jj < 320; jj += 32) {
    // --- S = Q @ K^T for 32 keys (two 16-key tiles) ---
    int k0r = jj + ln;        int k0c = k0r > 290 ? 290 : k0r;
    int k1r = jj + 16 + ln;   int k1c = k1r > 290 ? 290 : k1r;
    const _Float16* kp0 = qkv + base + (size_t)k0c * 1152 + DEMB + head * HDIM + grp * 16;
    const _Float16* kp1 = qkv + base + (size_t)k1c * 1152 + DEMB + head * HDIM + grp * 16;
    v16h b0lo = ldfrag16(kp0), b0hi = ldfrag16(kp0 + 32);
    v16h b1lo = ldfrag16(kp1), b1hi = ldfrag16(kp1 + 32);

    v8f s0 = zero, s1 = zero;
    s0 = wmma32(aq0, b0lo, s0);  s0 = wmma32(aq1, b0hi, s0);
    s1 = wmma32(aq0, b1lo, s1);  s1 = wmma32(aq1, b1hi, s1);

    // --- coalesced V row load (lane -> key jj+lane) ---
    int vk = jj + lane;  if (vk > 290) vk = 290;
    const _Float16* vp = qkv + base + (size_t)vk * 1152 + 2 * DEMB + head * HDIM;
    v8h vr0 = *(const v8h*)(vp +  0), vr1 = *(const v8h*)(vp +  8);
    v8h vr2 = *(const v8h*)(vp + 16), vr3 = *(const v8h*)(vp + 24);
    v8h vr4 = *(const v8h*)(vp + 32), vr5 = *(const v8h*)(vp + 40);
    v8h vr6 = *(const v8h*)(vp + 48), vr7 = *(const v8h*)(vp + 56);

    // --- masked online softmax over these 32 keys ---
    float p0[8], p1[8];
    #pragma unroll
    for (int i = 0; i < 8; ++i) {
      int qg = qb * 16 + grp * 8 + i;
      float v0 = s0[i], v1 = s1[i];
      bool ro = (qg == 290);
      if (k0r > 290 || (ro && k0r != 290)) v0 = NEG;
      if (k1r > 290 || (ro && k1r != 290)) v1 = NEG;
      float mx = fmaxf(v0, v1);
      #pragma unroll
      for (int off = 8; off > 0; off >>= 1)
        mx = fmaxf(mx, __shfl_xor(mx, off, 16));
      float nm = fmaxf(m[i], mx);
      sf[i] = __expf(m[i] - nm);
      m[i]  = nm;
      p0[i] = __expf(v0 - nm);
      p1[i] = __expf(v1 - nm);
      float ps = p0[i] + p1[i];
      #pragma unroll
      for (int off = 8; off > 0; off >>= 1)
        ps += __shfl_xor(ps, off, 16);
      lsum[i] = lsum[i] * sf[i] + ps;
    }

    // --- stage P and transposed V in LDS ---
    __syncthreads();
    #pragma unroll
    for (int i = 0; i < 8; ++i) {
      int row = grp * 8 + i;
      Pl[row * 32 + ln]      = (_Float16)p0[i];
      Pl[row * 32 + 16 + ln] = (_Float16)p1[i];
    }
    #pragma unroll
    for (int d = 0; d < 8; ++d) {
      VsT[(d +  0) * 40 + lane] = vr0[d];
      VsT[(d +  8) * 40 + lane] = vr1[d];
      VsT[(d + 16) * 40 + lane] = vr2[d];
      VsT[(d + 24) * 40 + lane] = vr3[d];
      VsT[(d + 32) * 40 + lane] = vr4[d];
      VsT[(d + 40) * 40 + lane] = vr5[d];
      VsT[(d + 48) * 40 + lane] = vr6[d];
      VsT[(d + 56) * 40 + lane] = vr7[d];
    }
    __syncthreads();

    v16h pa = ldfragA(&Pl[ln * 32 + grp * 16]);

    // --- O = O*sf + P @ V  (4 d-tiles of 16) ---
    #pragma unroll
    for (int dt = 0; dt < 4; ++dt) {
      v16h bv = ldfrag16(&VsT[(dt * 16 + ln) * 40 + grp * 16]);
      v8f ca = (dt == 0) ? acc0 : (dt == 1) ? acc1 : (dt == 2) ? acc2 : acc3;
      #pragma unroll
      for (int i = 0; i < 8; ++i) ca[i] *= sf[i];
      ca = wmma32(pa, bv, ca);
      if (dt == 0) acc0 = ca; else if (dt == 1) acc1 = ca;
      else if (dt == 2) acc2 = ca; else acc3 = ca;
    }
  }

  // store (normalize with hardware rcp)
  float inv[8];
  #pragma unroll
  for (int i = 0; i < 8; ++i) inv[i] = __builtin_amdgcn_rcpf(lsum[i]);
  #pragma unroll
  for (int dt = 0; dt < 4; ++dt) {
    v8f ca = (dt == 0) ? acc0 : (dt == 1) ? acc1 : (dt == 2) ? acc2 : acc3;
    #pragma unroll
    for (int i = 0; i < 8; ++i) {
      int qg = qb * 16 + grp * 8 + i;
      if (qg <= 290)
        out[(size_t)(b * TOK + qg) * DEMB + head * HDIM + dt * 16 + ln] =
            (_Float16)(ca[i] * inv[i]);
    }
  }
}

// ---------------------------------------------------------------------------
// LayerNorm: one wave32 per row, shuffle reductions, float4 loads.
// ---------------------------------------------------------------------------
__global__ void __launch_bounds__(32)
ln_kernel(const float* __restrict__ x, const float* __restrict__ w,
          const float* __restrict__ bb, _Float16* __restrict__ out) {
  int row = blockIdx.x, ln = threadIdx.x;
  const float* xr = x + (size_t)row * DEMB;
  float4 v[3];
  float s = 0.f;
  #pragma unroll
  for (int j = 0; j < 3; ++j) {
    v[j] = *(const float4*)&xr[ln * 4 + j * 128];
    s += v[j].x + v[j].y + v[j].z + v[j].w;
  }
  #pragma unroll
  for (int off = 16; off > 0; off >>= 1) s += __shfl_xor(s, off, 32);
  float mean = s * (1.f / DEMB);
  float sq = 0.f;
  #pragma unroll
  for (int j = 0; j < 3; ++j) {
    v[j].x -= mean; v[j].y -= mean; v[j].z -= mean; v[j].w -= mean;
    sq += v[j].x * v[j].x + v[j].y * v[j].y + v[j].z * v[j].z + v[j].w * v[j].w;
  }
  #pragma unroll
  for (int off = 16; off > 0; off >>= 1) sq += __shfl_xor(sq, off, 32);
  float rs = rsqrtf(sq * (1.f / DEMB) + 1e-5f);
  _Float16* orow = out + (size_t)row * DEMB;
  #pragma unroll
  for (int j = 0; j < 3; ++j) {
    int c = ln * 4 + j * 128;
    float4 wv = *(const float4*)&w[c];
    float4 bv = *(const float4*)&bb[c];
    orow[c + 0] = (_Float16)(v[j].x * rs * wv.x + bv.x);
    orow[c + 1] = (_Float16)(v[j].y * rs * wv.y + bv.y);
    orow[c + 2] = (_Float16)(v[j].z * rs * wv.z + bv.z);
    orow[c + 3] = (_Float16)(v[j].w * rs * wv.w + bv.w);
  }
}

// Final LN on readout rows only; fp32 output.
__global__ void __launch_bounds__(32)
final_ln_kernel(const float* __restrict__ x, const float* __restrict__ w,
                const float* __restrict__ bb, float* __restrict__ out) {
  int b = blockIdx.x, ln = threadIdx.x;
  const float* xr = x + (size_t)(b * TOK + 290) * DEMB;
  float4 v[3];
  float s = 0.f;
  #pragma unroll
  for (int j = 0; j < 3; ++j) {
    v[j] = *(const float4*)&xr[ln * 4 + j * 128];
    s += v[j].x + v[j].y + v[j].z + v[j].w;
  }
  #pragma unroll
  for (int off = 16; off > 0; off >>= 1) s += __shfl_xor(s, off, 32);
  float mean = s * (1.f / DEMB);
  float sq = 0.f;
  #pragma unroll
  for (int j = 0; j < 3; ++j) {
    v[j].x -= mean; v[j].y -= mean; v[j].z -= mean; v[j].w -= mean;
    sq += v[j].x * v[j].x + v[j].y * v[j].y + v[j].z * v[j].z + v[j].w * v[j].w;
  }
  #pragma unroll
  for (int off = 16; off > 0; off >>= 1) sq += __shfl_xor(sq, off, 32);
  float rs = rsqrtf(sq * (1.f / DEMB) + 1e-5f);
  float* orow = out + (size_t)b * DEMB;
  #pragma unroll
  for (int j = 0; j < 3; ++j) {
    int c = ln * 4 + j * 128;
    float4 wv = *(const float4*)&w[c];
    float4 bv = *(const float4*)&bb[c];
    float4 o;
    o.x = v[j].x * rs * wv.x + bv.x;
    o.y = v[j].y * rs * wv.y + bv.y;
    o.z = v[j].z * rs * wv.z + bv.z;
    o.w = v[j].w * rs * wv.w + bv.w;
    *(float4*)&orow[c] = o;
  }
}

// ---------------------------------------------------------------------------
// Tokenizers -> residual stream h (fp32 [37248 x 384]); token = t*145 + {0:
// state, 1+p: img patch p}; readout at 290. Adds temporal + positional embeds.
// ---------------------------------------------------------------------------
__global__ void __launch_bounds__(128)
img_tok_kernel(const float* __restrict__ images, const float* __restrict__ Wp,
               const float* __restrict__ bp, const float* __restrict__ temb,
               const float* __restrict__ pos, float* __restrict__ h) {
  int id = blockIdx.x;
  int patch = id % 144;  int bt = id / 144;
  int t = bt & 1, b = bt >> 1;
  int py = patch / 12, px = patch % 12;
  __shared__ float pv[192];
  for (int idx = threadIdx.x; idx < 192; idx += 128) {
    int c = idx % 3, rem = idx / 3;
    int ppx = rem % 8, ppy = rem / 8;
    pv[idx] = images[(((size_t)bt * 3 + c) * 96 + py * 8 + ppy) * 96 + px * 8 + ppx];
  }
  __syncthreads();
  int tok = t * 145 + 1 + patch;
  float* hr = h + (size_t)(b * TOK + tok) * DEMB;
  #pragma unroll
  for (int j = 0; j < 3; ++j) {
    int d = threadIdx.x + j * 128;
    float val = bp[d] + temb[t * DEMB + d] + pos[(size_t)tok * DEMB + d];
    for (int i = 0; i < 192; ++i) val += pv[i] * Wp[i * DEMB + d];
    hr[d] = val;
  }
}

__global__ void __launch_bounds__(128)
state_tok_kernel(const float* __restrict__ st, const float* __restrict__ w1,
                 const float* __restrict__ b1, const float* __restrict__ w2,
                 const float* __restrict__ b2, const float* __restrict__ temb,
                 const float* __restrict__ pos, float* __restrict__ h) {
  int bt = blockIdx.x;
  int t = bt & 1, b = bt >> 1;
  __shared__ float hid[32];
  if (threadIdx.x < 32) {
    float s = b1[threadIdx.x]
            + st[bt * 2 + 0] * w1[0 * 32 + threadIdx.x]
            + st[bt * 2 + 1] * w1[1 * 32 + threadIdx.x];
    hid[threadIdx.x] = fmaxf(s, 0.f);
  }
  __syncthreads();
  int tok = t * 145;
  float* hr = h + (size_t)(b * TOK + tok) * DEMB;
  #pragma unroll
  for (int j = 0; j < 3; ++j) {
    int d = threadIdx.x + j * 128;
    float val = b2[d] + temb[t * DEMB + d] + pos[(size_t)tok * DEMB + d];
    for (int i = 0; i < 32; ++i) val += hid[i] * w2[i * DEMB + d];
    hr[d] = val;
  }
}

__global__ void __launch_bounds__(128)
readout_tok_kernel(const float* __restrict__ ro, const float* __restrict__ pos,
                   float* __restrict__ h) {
  int b = blockIdx.x;
  float* hr = h + (size_t)(b * TOK + 290) * DEMB;
  #pragma unroll
  for (int j = 0; j < 3; ++j) {
    int d = threadIdx.x + j * 128;
    hr[d] = ro[d] + pos[(size_t)290 * DEMB + d];
  }
}

// ---------------------------------------------------------------------------
extern "C" void kernel_launch(void* const* d_in, const int* in_sizes, int n_in,
                              void* d_out, int out_size, void* d_ws, size_t ws_size,
                              hipStream_t stream) {
  const float* images      = (const float*)d_in[0];
  const float* agent_state = (const float*)d_in[1];
  const float* img_proj_w  = (const float*)d_in[2];
  const float* img_proj_b  = (const float*)d_in[3];
  const float* img_temb    = (const float*)d_in[4];
  const float* st_w1       = (const float*)d_in[5];
  const float* st_b1       = (const float*)d_in[6];
  const float* st_w2       = (const float*)d_in[7];
  const float* st_b2       = (const float*)d_in[8];
  const float* st_temb     = (const float*)d_in[9];
  const float* readout     = (const float*)d_in[10];
  const float* pos_embed   = (const float*)d_in[11];
  const float* ln1_w       = (const float*)d_in[12];
  const float* ln1_b       = (const float*)d_in[13];
  const float* attn_w      = (const float*)d_in[14];
  const float* attn_b      = (const float*)d_in[15];
  const float* attn_proj_w = (const float*)d_in[16];
  const float* attn_proj_b = (const float*)d_in[17];
  const float* ln2_w       = (const float*)d_in[18];
  const float* ln2_b       = (const float*)d_in[19];
  const float* fc_w        = (const float*)d_in[20];
  const float* fc_b        = (const float*)d_in[21];
  const float* fc2_w       = (const float*)d_in[22];
  const float* fc2_b       = (const float*)d_in[23];
  const float* lnf_w       = (const float*)d_in[24];
  const float* lnf_b       = (const float*)d_in[25];

  char* ws = (char*)d_ws;
  size_t off = 0;
  float*    h     = (float*)(ws + off);    off += (size_t)MROWS * DEMB * 4;   // residual
  _Float16* ln16  = (_Float16*)(ws + off); off += (size_t)MROWS * DEMB * 2;   // LN out
  _Float16* big16 = (_Float16*)(ws + off); off += (size_t)MROWS * 1536 * 2;   // qkv / MLP mid
  _Float16* ao16  = (_Float16*)(ws + off); off += (size_t)MROWS * DEMB * 2;   // attn out
  (void)off; (void)ws_size; (void)n_in; (void)in_sizes; (void)out_size;

  // --- tokenize into h ---
  img_tok_kernel<<<NBATCH * 2 * 144, 128, 0, stream>>>(images, img_proj_w, img_proj_b,
                                                       img_temb, pos_embed, h);
  state_tok_kernel<<<NBATCH * 2, 128, 0, stream>>>(agent_state, st_w1, st_b1, st_w2, st_b2,
                                                   st_temb, pos_embed, h);
  readout_tok_kernel<<<NBATCH, 128, 0, stream>>>(readout, pos_embed, h);

  // --- transformer layers ---
  for (int L = 0; L < NLAYER; ++L) {
    ln_kernel<<<MROWS, 32, 0, stream>>>(h, ln1_w + L * DEMB, ln1_b + L * DEMB, ln16);
    gemm_kernel<<<dim3(1152 / 64, MROWS / 128), 256, 0, stream>>>(
        ln16, attn_w + (size_t)L * DEMB * 1152, attn_b + L * 1152,
        nullptr, big16, DEMB, 1152, 0);
    attn_kernel<<<NBATCH * NHEAD * 19, 32, 0, stream>>>(big16, ao16);
    gemm_kernel<<<dim3(DEMB / 64, MROWS / 128), 256, 0, stream>>>(
        ao16, attn_proj_w + (size_t)L * DEMB * DEMB, attn_proj_b + L * DEMB,
        h, nullptr, DEMB, DEMB, 2);
    ln_kernel<<<MROWS, 32, 0, stream>>>(h, ln2_w + L * DEMB, ln2_b + L * DEMB, ln16);
    gemm_kernel<<<dim3(1536 / 64, MROWS / 128), 256, 0, stream>>>(
        ln16, fc_w + (size_t)L * DEMB * 1536, fc_b + L * 1536,
        nullptr, big16, DEMB, 1536, 1);
    gemm_kernel<<<dim3(DEMB / 64, MROWS / 128), 256, 0, stream>>>(
        big16, fc2_w + (size_t)L * 1536 * DEMB, fc2_b + L * DEMB,
        h, nullptr, 1536, DEMB, 2);
  }

  final_ln_kernel<<<NBATCH, 32, 0, stream>>>(h, lnf_w, lnf_b, (float*)d_out);
}
